// TimestepAwarePTQLinear_34179349741725
// MI455X (gfx1250) — matrix operand order repaired and checked by reading, not compiled, not run on hardware
//
#include <hip/hip_runtime.h>
#include <hip/hip_bf16.h>

// ---------------------------------------------------------------------------
// Problem constants (all divisions below are exact)
// ---------------------------------------------------------------------------
constexpr int M = 4 * 4096;   // B*S = 16384 rows
constexpr int D = 2048;       // K
constexpr int O = 2048;       // N
constexpr int HBLK = 128;     // hadamard block
constexpr int NHB = D / HBLK; // 16 hadamard blocks
constexpr int NRANK = 36;     // 32 lora + 4 delta ranks

typedef _Float16 half16 __attribute__((ext_vector_type(16)));
typedef _Float16 half8  __attribute__((ext_vector_type(8)));
typedef float    float8 __attribute__((ext_vector_type(8)));
typedef unsigned u32x4  __attribute__((ext_vector_type(4)));
typedef int      i32x4  __attribute__((ext_vector_type(4)));
typedef int      i32x8  __attribute__((ext_vector_type(8)));

union V16 {
    half16   v;
    _Float16 e[16];
};

// ---------------------------------------------------------------------------
// TDM: async 2D tile DMA between global and LDS (Tensor Data Mover).
// D# packing per CDNA5 ISA: count=1, type=2,
// huge tensor dims (never OOB), optional LDS padding on loads.
// ---------------------------------------------------------------------------
__device__ inline void tdm_tile(bool is_store, unsigned lds_byte_addr,
                                const void* gaddr, unsigned data_size_code,
                                unsigned tile_x, unsigned tile_rows,
                                unsigned stride_elems, unsigned pad_en,
                                unsigned pad_interval_code,
                                unsigned pad_amount_code) {
    const unsigned long long ga = (unsigned long long)(size_t)gaddr;
    u32x4 g0;
    g0[0] = 1u;                                 // count=1, user desc, no gather
    g0[1] = lds_byte_addr;                      // lds_addr
    g0[2] = (unsigned)ga;                       // global_addr[31:0]
    g0[3] = (unsigned)(ga >> 32) | (2u << 30);  // global_addr[56:32] | type=2
    const unsigned tdim0 = 1u << 30;            // huge dims: never OOB
    const unsigned tdim1 = 1u << 30;
    i32x8 g1;
    g1[0] = (int)((data_size_code << 16) | (pad_en << 20)
                | (pad_interval_code << 22) | (pad_amount_code << 25));
    g1[1] = (int)((tdim0 & 0xFFFFu) << 16);                    // dim0 lo
    g1[2] = (int)((tdim0 >> 16) | ((tdim1 & 0xFFFFu) << 16));  // dim0 hi|dim1 lo
    g1[3] = (int)((tdim1 >> 16) | (tile_x << 16));             // dim1 hi|tile_dim0
    g1[4] = (int)tile_rows;                     // tile_dim1 (tile_dim2 = 0)
    g1[5] = (int)stride_elems;                  // tensor_dim0_stride (elements)
    g1[6] = 0;
    g1[7] = 0;
    const i32x4 z4 = {0, 0, 0, 0};
#if __clang_major__ >= 23
    const i32x8 z8 = {0, 0, 0, 0, 0, 0, 0, 0};
    if (is_store) __builtin_amdgcn_tensor_store_from_lds(g0, g1, z4, z4, z8, 0);
    else          __builtin_amdgcn_tensor_load_to_lds(g0, g1, z4, z4, z8, 0);
#else
    if (is_store) __builtin_amdgcn_tensor_store_from_lds(g0, g1, z4, z4, 0);
    else          __builtin_amdgcn_tensor_load_to_lds(g0, g1, z4, z4, 0);
#endif
}

// ---------------------------------------------------------------------------
// Kernel 0: convert w_quantized f32[O,D] -> f16[O,D] row-major
// ---------------------------------------------------------------------------
__global__ __launch_bounds__(256)
void wcvt_kernel(const float* __restrict__ w, _Float16* __restrict__ w16) {
    size_t i = ((size_t)blockIdx.x * 256 + threadIdx.x) * 8;
    float4 a = *(const float4*)&w[i];
    float4 b = *(const float4*)&w[i + 4];
    half8 h;
    h[0] = (_Float16)a.x; h[1] = (_Float16)a.y;
    h[2] = (_Float16)a.z; h[3] = (_Float16)a.w;
    h[4] = (_Float16)b.x; h[5] = (_Float16)b.y;
    h[6] = (_Float16)b.z; h[7] = (_Float16)b.w;
    *(half8*)&w16[i] = h;
}

// ---------------------------------------------------------------------------
// Kernel 1: blockHadamard (WMMA) + smooth + int4 fake-quant -> xq (f16),
//           plus r = x_smooth @ [lora_a; delta_a[g]]^T (WMMA, ranks padded
//           36->48 with zero LDS rows). x_smooth never leaves the chip.
// Bulk bytes ride the TDM: x tiles DMA'd in (double-buffered, prefetching
// block jb+1 during block jb's compute), x_q tiles DMA'd out per wave.
// Workgroup: 128 threads (4 waves), owns 64 rows, loops over 16 D-blocks.
// ---------------------------------------------------------------------------
__global__ __launch_bounds__(128)
void prep_kernel(const float* __restrict__ x,
                 const float* __restrict__ H,
                 const float* __restrict__ smooth,
                 const float* __restrict__ lora_a,
                 const float* __restrict__ delta_a,
                 const int*   __restrict__ gptr,
                 _Float16*    __restrict__ xq,
                 float*       __restrict__ rbuf) {
    __shared__ _Float16 Hs[128][136];    // H in f16 (symmetric: row == column)
    __shared__ float    Xs[2][64][132];  // raw x tiles (TDM pad -> 132 pitch)
    __shared__ _Float16 Ss[64][136];     // x_smooth tile (f16)
    __shared__ _Float16 Aa[48][136];     // A_cat slice (f16), rows 36..47 zero
    __shared__ _Float16 Qs[64][128];     // packed x_q tile for TDM store

    const int tid  = threadIdx.x;
    const int lane = tid & 31;
    const int wave = tid >> 5;
    const int m0   = blockIdx.x * 64;
    const int g    = *gptr;

    // Kick off the first x-tile DMA immediately (overlaps H staging).
    // f32 tile: 128 elems/row, 64 rows, pad 4 dwords after every 128 dwords.
    if (wave == 0)
        tdm_tile(false, (unsigned)(size_t)&Xs[0][0][0], &x[(size_t)m0 * D],
                 2u, 128u, 64u, (unsigned)D, 1u, 6u, 3u);

    // Load Hadamard block once (f32 -> f16)
    for (int i = tid; i < 128 * 128 / 4; i += 128) {
        const float4 hv = ((const float4*)H)[i];
        const int lin = i * 4;
        const int row = lin >> 7;
        const int col = lin & 127;
        Hs[row][col + 0] = (_Float16)hv.x;
        Hs[row][col + 1] = (_Float16)hv.y;
        Hs[row][col + 2] = (_Float16)hv.z;
        Hs[row][col + 3] = (_Float16)hv.w;
    }
    // Zero the rank-padding rows of Aa once (rows 36..47 never rewritten)
    for (int i = tid; i < 12 * 136; i += 128)
        Aa[36 + i / 136][i % 136] = (_Float16)0.f;

    const int mrow  = wave * 16;                 // wave's row slice
    const int am    = lane & 15;                 // fragment row/col selector
    const int koff  = (lane >> 4) * 8;           // A-frag K offset
    const int bk    = (lane >> 4) * 16;          // unused; B frags load with koff too
    const int rbase = mrow + ((lane >> 4) << 3); // C-frag row base

    float8 racc[3] = {};                         // rank C tiles (48 cols)

    for (int jb = 0; jb < NHB; ++jb) {
        const int jcol = jb * HBLK;
        // Own TDM ops retired: x(jb) landed in Xs[jb&1] (wave0) and the
        // previous Qs store drained (all waves) -> Qs reusable.
        __builtin_amdgcn_s_wait_tensorcnt(0);
        __syncthreads();   // publish TDM-written Xs; prior Aa/Xs readers done

        // Prefetch next x tile into the other buffer (its previous readers
        // finished before the barrier above).
        if (wave == 0 && jb + 1 < NHB)
            tdm_tile(false, (unsigned)(size_t)&Xs[(jb + 1) & 1][0][0],
                     &x[(size_t)m0 * D + (size_t)(jb + 1) * HBLK],
                     2u, 128u, 64u, (unsigned)D, 1u, 6u, 3u);

        // Stage A_cat slice [36][128] as f16
        for (int i = tid; i < 36 * 128; i += 128) {
            const int rr  = i >> 7;
            const int col = i & 127;
            const float v = (rr < 32)
                ? lora_a[(size_t)rr * D + jcol + col]
                : delta_a[(size_t)(g * 4 + (rr - 32)) * D + jcol + col];
            Aa[rr][col] = (_Float16)v;
        }
        __syncthreads();

        const float (*Xb)[132] = Xs[jb & 1];

        // A fragments of raw x (16 rows x 32 K each), reused for 8 N tiles
        V16 afrag[4];
#pragma unroll
        for (int c = 0; c < 4; ++c) {
            const float* src = &Xb[mrow + am][c * 32 + koff];
#pragma unroll
            for (int e = 0; e < 8; ++e) {
                afrag[c].e[e]     = (_Float16)src[e];
                afrag[c].e[e + 8] = (_Float16)src[16 + e];
            }
        }

#pragma unroll
        for (int nt = 0; nt < 8; ++nt) {
            float8 acc = {};
#pragma unroll
            for (int c = 0; c < 4; ++c) {
                V16 bfrag;
                const _Float16* hsrc = &Hs[nt * 16 + am][c * 32 + koff]; // H symm.
                *(half8*)&bfrag.e[0] = *(const half8*)&hsrc[0];
                *(half8*)&bfrag.e[8] = *(const half8*)&hsrc[16];
                acc = __builtin_amdgcn_wmma_f32_16x16x32_f16(
                    false, afrag[c].v, false, bfrag.v, (short)0, acc, false, false);
            }
            // smooth scale (per channel)
            const int col = jcol + nt * 16 + am;
            const float s = smooth[(size_t)g * D + col];
            float vals[8];
#pragma unroll
            for (int v = 0; v < 8; ++v) vals[v] = acc[v] * s;
            // publish x_smooth (f16) for the rank projection
#pragma unroll
            for (int v = 0; v < 8; ++v)
                Ss[rbase + v][nt * 16 + am] = (_Float16)vals[v];
            // int4 fake-quant: quant block (16 consecutive d) == one C tile's
            // 16-lane row group; xor 1,2,4,8 reduction stays inside the group.
#pragma unroll
            for (int v = 0; v < 8; ++v) {
                float a = fabsf(vals[v]);
                a = fmaxf(a, __shfl_xor(a, 1, 32));
                a = fmaxf(a, __shfl_xor(a, 2, 32));
                a = fmaxf(a, __shfl_xor(a, 4, 32));
                a = fmaxf(a, __shfl_xor(a, 8, 32));
                const float scale = fmaxf(a, 1e-12f) * (1.0f / 7.0f);
                float q = rintf(vals[v] / scale);
                q = fminf(fmaxf(q, -7.f), 7.f) * scale;
                Qs[rbase + v][nt * 16 + am] = (_Float16)q;   // packed tile
            }
        }

        // rank-36 projection via WMMA: racc += Ss(16x128) @ Aa^T (128x48)
        // (Ss rows are this wave's own -> no barrier needed before reading)
        V16 sfrag[4];
#pragma unroll
        for (int c = 0; c < 4; ++c) {
            const _Float16* s = &Ss[mrow + am][c * 32 + koff];
            *(half8*)&sfrag[c].e[0] = *(const half8*)&s[0];
            *(half8*)&sfrag[c].e[8] = *(const half8*)&s[16];
        }
#pragma unroll
        for (int nt = 0; nt < 3; ++nt) {
#pragma unroll
            for (int c = 0; c < 4; ++c) {
                V16 bfrag;
                const _Float16* a = &Aa[nt * 16 + am][c * 32 + koff];
                *(half8*)&bfrag.e[0] = *(const half8*)&a[0];
                *(half8*)&bfrag.e[8] = *(const half8*)&a[16];
                racc[nt] = __builtin_amdgcn_wmma_f32_16x16x32_f16(
                    false, sfrag[c].v, false, bfrag.v, (short)0,
                    racc[nt], false, false);
            }
        }

        // Flush this wave's quantized 16x128 slice to xq via TDM store.
        // TDM is unordered vs LDS ops -> drain our ds writes first.
        asm volatile("s_wait_dscnt 0" ::: "memory");
        tdm_tile(true, (unsigned)(size_t)&Qs[mrow][0],
                 &xq[(size_t)(m0 + mrow) * D + jcol],
                 1u, 128u, 16u, (unsigned)D, 0u, 0u, 0u);
    }

    // Store r (drop the 12 zero-padded rank columns)
#pragma unroll
    for (int nt = 0; nt < 3; ++nt) {
        const int rr = nt * 16 + am;
        if (rr < NRANK) {
#pragma unroll
            for (int v = 0; v < 8; ++v)
                rbuf[(size_t)(m0 + rbase + v) * NRANK + rr] = racc[nt][v];
        }
    }
}

// ---------------------------------------------------------------------------
// Kernel 2: out = xq @ w16^T (WMMA f16, f32 acc) + r @ B_cat^T + bias
// Block tile 128x128, 8 waves (2 M x 4 N), wave tile 64x32, K chunk 64,
// double-buffered LDS fed by the Tensor Data Mover (async DMA).
// ---------------------------------------------------------------------------
constexpr int KC = 64;

__global__ __launch_bounds__(256)
void gemm_kernel(const _Float16* __restrict__ xq,
                 const _Float16* __restrict__ w16,
                 const float*    __restrict__ rbuf,
                 const float*    __restrict__ lora_b,
                 const float*    __restrict__ delta_b,
                 const float*    __restrict__ bias,
                 const int*      __restrict__ gptr,
                 float*          __restrict__ out) {
    __shared__ _Float16 As[2][128][KC + 8];  // 72-half pitch (TDM pad writes this)
    __shared__ _Float16 Bs[2][128][KC + 8];

    const int tid  = threadIdx.x;
    const int lane = tid & 31;
    const int wave = tid >> 5;
    const int m0   = blockIdx.x * 128;
    const int n0   = blockIdx.y * 128;
    const int wm   = (wave >> 2) * 64;   // 2 waves along M
    const int wn   = (wave & 3) * 32;    // 4 waves along N
    const int g    = *gptr;

    float8 acc[4][2] = {};

    const int am   = lane & 15;
    const int koff = (lane >> 4) * 8;
    const int bk   = (lane >> 4) * 16;

    // wave0 DMAs the A tile, wave1 the B tile (TDM ignores EXEC; one op/wave)
    // f16 tile: 64 elems/row, 128 rows, pad 4 dwords after every 32 dwords.
    auto issueTile = [&](int it, int buf) {
        const int k0 = it * KC;
        if (wave == 0)
            tdm_tile(false, (unsigned)(size_t)&As[buf][0][0],
                     &xq[(size_t)m0 * D + k0], 1u, 64u, 128u, (unsigned)D,
                     1u, 4u, 3u);
        else if (wave == 1)
            tdm_tile(false, (unsigned)(size_t)&Bs[buf][0][0],
                     &w16[(size_t)n0 * D + k0], 1u, 64u, 128u, (unsigned)D,
                     1u, 4u, 3u);
    };

    auto compute = [&](int buf) {
#pragma unroll
        for (int ks = 0; ks < 2; ++ks) {
            V16 af[4];
#pragma unroll
            for (int mt = 0; mt < 4; ++mt) {
                const _Float16* s = &As[buf][wm + mt * 16 + am][ks * 32 + koff];
                *(half8*)&af[mt].e[0] = *(const half8*)&s[0];
                *(half8*)&af[mt].e[8] = *(const half8*)&s[16];
            }
            V16 bf[2];
#pragma unroll
            for (int nt = 0; nt < 2; ++nt) {
                const _Float16* s = &Bs[buf][wn + nt * 16 + am][ks * 32 + koff];
                *(half8*)&bf[nt].e[0] = *(const half8*)&s[0];
                *(half8*)&bf[nt].e[8] = *(const half8*)&s[16];
            }
#pragma unroll
            for (int mt = 0; mt < 4; ++mt)
#pragma unroll
                for (int nt = 0; nt < 2; ++nt)
                    acc[mt][nt] = __builtin_amdgcn_wmma_f32_16x16x32_f16(
                        false, af[mt].v, false, bf[nt].v, (short)0,
                        acc[mt][nt], false, false);
        }
    };

    const int NIT = D / KC;  // 32
    issueTile(0, 0);
    if (wave < 2) __builtin_amdgcn_s_wait_tensorcnt(0);
    __syncthreads();
    for (int it = 0; it < NIT; ++it) {
        if (it + 1 < NIT) issueTile(it + 1, (it + 1) & 1);  // DMA next chunk
        compute(it & 1);                                    // overlap compute
        if (it + 1 < NIT && wave < 2) __builtin_amdgcn_s_wait_tensorcnt(0);
        __syncthreads();
    }

    // Epilogue: += r[m,:36] . Bcat[o,:36] + bias[o]
    const int cl = lane & 15;
    const int ch = (lane >> 4) << 3;
    float lb[2][36];
    float bo[2];
#pragma unroll
    for (int nt = 0; nt < 2; ++nt) {
        const int o = n0 + wn + nt * 16 + cl;
#pragma unroll
        for (int j = 0; j < 32; ++j) lb[nt][j] = lora_b[(size_t)o * 32 + j];
#pragma unroll
        for (int j = 0; j < 4; ++j)
            lb[nt][32 + j] = delta_b[((size_t)g * O + o) * 4 + j];
        bo[nt] = bias[o];
    }
#pragma unroll
    for (int mt = 0; mt < 4; ++mt) {
#pragma unroll
        for (int v = 0; v < 8; ++v) {
            const int m = m0 + wm + mt * 16 + ch + v;
            const float* rp = &rbuf[(size_t)m * NRANK];
            float rrow[36];
#pragma unroll
            for (int j = 0; j < 36; ++j) rrow[j] = rp[j];
#pragma unroll
            for (int nt = 0; nt < 2; ++nt) {
                float sum = acc[mt][nt][v] + bo[nt];
#pragma unroll
                for (int j = 0; j < 36; ++j) sum += rrow[j] * lb[nt][j];
                const int o = n0 + wn + nt * 16 + cl;
                out[(size_t)m * O + o] = sum;
            }
        }
    }
}

// ---------------------------------------------------------------------------
// Launch
// ---------------------------------------------------------------------------
extern "C" void kernel_launch(void* const* d_in, const int* in_sizes, int n_in,
                              void* d_out, int out_size, void* d_ws, size_t ws_size,
                              hipStream_t stream) {
    const float* x       = (const float*)d_in[0];
    const float* H       = (const float*)d_in[1];
    const float* smooth  = (const float*)d_in[2];
    const float* w       = (const float*)d_in[3];
    const float* lora_a  = (const float*)d_in[4];
    const float* lora_b  = (const float*)d_in[5];
    const float* delta_a = (const float*)d_in[6];
    const float* delta_b = (const float*)d_in[7];
    const float* bias    = (const float*)d_in[8];
    const int*   gptr    = (const int*)d_in[9];
    float* out = (float*)d_out;

    // Workspace: xq f16 [M,D] (64 MiB) | w16 f16 [O,D] (8 MiB) | r f32 [M,36]
    char* ws = (char*)d_ws;
    _Float16* xq  = (_Float16*)ws;
    _Float16* w16 = (_Float16*)(ws + (size_t)M * D * sizeof(_Float16));
    float*    rb  = (float*)(ws + (size_t)M * D * sizeof(_Float16)
                                + (size_t)O * D * sizeof(_Float16));

    wcvt_kernel<<<(O * D) / (256 * 8), 256, 0, stream>>>(w, w16);
    prep_kernel<<<M / 64, 128, 0, stream>>>(x, H, smooth, lora_a, delta_a,
                                            gptr, xq, rb);
    gemm_kernel<<<dim3(M / 128, O / 128), 256, 0, stream>>>(
        xq, w16, rb, lora_b, delta_b, bias, gptr, out);
}
